// MultiHeadAttention_91319594647721
// MI455X (gfx1250) — compile-verified, matrix operand order checked by reading, not run on hardware
//
#include <hip/hip_runtime.h>
#include <stdint.h>

// ---------------------------------------------------------------------------
// MultiHeadAttention block for MI455X (gfx1250, wave32, WMMA bf16)
// B=8, L=2048, D=1024, H=4, DK=256
// ---------------------------------------------------------------------------

typedef __attribute__((ext_vector_type(16))) __bf16 v16bf;
typedef __attribute__((ext_vector_type(8)))  __bf16 v8bf;
typedef __attribute__((ext_vector_type(4)))  __bf16 v4bf;
typedef __attribute__((ext_vector_type(8)))  float  v8f;
typedef __attribute__((ext_vector_type(4)))  float  v4f;

#define B_   8
#define L_   2048
#define D_   1024
#define H_   4
#define DK_  256
#define M_   (B_ * L_)          // 16384 tokens
#define EPS_ 1e-5f

// f32 -> bf16 (hardware cvt, RNE)
__device__ __forceinline__ __bf16 f2bf(float f) { return (__bf16)f; }

// pack 8 contiguous f32 (two v4f) into one v8bf via native conversions
__device__ __forceinline__ v8bf pack8(v4f f0, v4f f1) {
  v4bf a = __builtin_convertvector(f0, v4bf);
  v4bf b = __builtin_convertvector(f1, v4bf);
  return __builtin_shufflevector(a, b, 0, 1, 2, 3, 4, 5, 6, 7);
}

__device__ __forceinline__ v8f vzero8() {
  v8f z;
#pragma unroll
  for (int i = 0; i < 8; ++i) z[i] = 0.f;
  return z;
}

union ABFrag { v16bf v; v8bf h[2]; };

__device__ __forceinline__ v8f wmma_bf16(v16bf a, v16bf b, v8f c) {
  // D = A(16x32 bf16) * B(32x16 bf16) + C(16x16 f32)
  return __builtin_amdgcn_wmma_f32_16x16x32_bf16(
      /*neg_a=*/false, a, /*neg_b=*/false, b,
      /*c_mod=*/(short)0, c, /*reuse_a=*/false, /*reuse_b=*/false);
}

// ---------------------------------------------------------------------------
// Kernel 1: QKV projections.  C[m,n] = sum_k A[m,k] * W[n,k] + b[n]
// z=0: q->QP  (row major bf16 [M,D]), pre-scaled by 1/sqrt(DK)
// z=1: k->KP  (row major bf16 [M,D])
// z=2: v->VPT (transposed bf16 [B, D(=H*DK), L]) for P*V B-fragments
// Block tile 128x64, K-step 64, 8 waves each doing 32x32 (2x2 WMMA frags)
// ---------------------------------------------------------------------------
__global__ __launch_bounds__(256) void qkv_proj_kernel(
    const float* __restrict__ qin, const float* __restrict__ kin,
    const float* __restrict__ vin,
    const float* __restrict__ Wq, const float* __restrict__ bq,
    const float* __restrict__ Wk, const float* __restrict__ bk,
    const float* __restrict__ Wv, const float* __restrict__ bv,
    __bf16* __restrict__ QP, __bf16* __restrict__ KP,
    __bf16* __restrict__ VPT) {
  const int z = blockIdx.z;
  const float* A    = (z == 0) ? qin : (z == 1) ? kin : vin;
  const float* W    = (z == 0) ? Wq  : (z == 1) ? Wk  : Wv;
  const float* bias = (z == 0) ? bq  : (z == 1) ? bk  : bv;
  const float scale = (z == 0) ? 0.0625f : 1.0f;  // fold 1/sqrt(256) into Q

  const int m0   = blockIdx.y * 128;
  const int n0   = blockIdx.x * 64;
  const int tid  = threadIdx.x;
  const int wave = tid >> 5;
  const int lane = tid & 31;
  const int half = lane >> 4;
  const int lrow = lane & 15;
  const int wm   = wave & 3;   // 4 wave-rows of 32
  const int wn   = wave >> 2;  // 2 wave-cols of 32
  const int c0   = half ? 8 : 0;

  __shared__ __align__(16) __bf16 As[128][64];  // 16 KB
  __shared__ __align__(16) __bf16 Bs[64][64];   //  8 KB

  v8f acc[2][2];
#pragma unroll
  for (int mr = 0; mr < 2; ++mr)
#pragma unroll
    for (int nr = 0; nr < 2; ++nr) acc[mr][nr] = vzero8();

  for (int k0 = 0; k0 < D_; k0 += 64) {
    // stage A tile 128x64: 4 chunks of 8 f32 -> 8 bf16 per thread
#pragma unroll
    for (int i = 0; i < 4; ++i) {
      const int e = (tid + i * 256) * 8;   // element index in tile
      const int r = e >> 6, c = e & 63;
      const v4f* src = (const v4f*)&A[(size_t)(m0 + r) * D_ + k0 + c];
      *(v8bf*)&As[r][c] = pack8(src[0], src[1]);
    }
    // stage W tile 64x64: 2 chunks per thread
#pragma unroll
    for (int i = 0; i < 2; ++i) {
      const int e = (tid + i * 256) * 8;
      const int r = e >> 6, c = e & 63;
      const v4f* src = (const v4f*)&W[(size_t)(n0 + r) * D_ + k0 + c];
      *(v8bf*)&Bs[r][c] = pack8(src[0], src[1]);
    }
    if (k0 + 64 < D_)  // -> global_prefetch_b8 for next K tile
      __builtin_prefetch(&A[(size_t)(m0 + (tid >> 1)) * D_ + k0 + 64], 0, 1);
    __syncthreads();

#pragma unroll
    for (int ks = 0; ks < 2; ++ks) {      // two 32-wide WMMA K-slices
      const int kc = ks * 32 + c0;
      ABFrag af[2], bf_[2];
#pragma unroll
      for (int mr = 0; mr < 2; ++mr) {
        const int ar = wm * 32 + mr * 16 + lrow;
        af[mr].h[0] = *(const v8bf*)&As[ar][kc];
        af[mr].h[1] = *(const v8bf*)&As[ar][kc + 16];
      }
#pragma unroll
      for (int nr = 0; nr < 2; ++nr) {
        const int br = wn * 32 + nr * 16 + lrow;
        bf_[nr].h[0] = *(const v8bf*)&Bs[br][kc];
        bf_[nr].h[1] = *(const v8bf*)&Bs[br][kc + 16];
      }
#pragma unroll
      for (int mr = 0; mr < 2; ++mr)
#pragma unroll
        for (int nr = 0; nr < 2; ++nr)
          acc[mr][nr] = wmma_bf16(af[mr].v, bf_[nr].v, acc[mr][nr]);
    }
    __syncthreads();
  }

  // epilogue: (+bias)*scale, store
#pragma unroll
  for (int nr = 0; nr < 2; ++nr) {
    const int n = n0 + wn * 32 + nr * 16 + lrow;
    const float bv_ = bias[n];
#pragma unroll
    for (int mr = 0; mr < 2; ++mr) {
      const int mbase = m0 + wm * 32 + mr * 16 + half * 8;  // 8 consecutive tokens
      if (z == 2) {
        // transposed: VPT[(b*D + n)*L + l], 8 consecutive l -> one b128 store
        const int bidx = mbase >> 11;        // / L_
        const int l    = mbase & (L_ - 1);
        v8bf pack;
#pragma unroll
        for (int r = 0; r < 8; ++r) pack[r] = f2bf(acc[mr][nr][r] + bv_);
        *(v8bf*)(VPT + ((size_t)(bidx * D_ + n)) * L_ + l) = pack;
      } else {
        __bf16* OUT = (z == 0) ? QP : KP;
#pragma unroll
        for (int r = 0; r < 8; ++r)
          OUT[(size_t)(mbase + r) * D_ + n] =
              f2bf((acc[mr][nr][r] + bv_) * scale);
      }
    }
  }
}

// ---------------------------------------------------------------------------
// Kernel 2: flash attention. 4 waves/block, each wave owns one 16-row Q tile
// and a private LDS slice (DS ops are in-order within a wave -> no barriers).
// S = (Q/16)*K^T, mask -> -inf, online softmax (NaN-fix), O += P*V.
// ---------------------------------------------------------------------------
__global__ __launch_bounds__(128) void flash_attn_kernel(
    const __bf16* __restrict__ QP, const __bf16* __restrict__ KP,
    const __bf16* __restrict__ VPT, const unsigned char* __restrict__ mask,
    __bf16* __restrict__ O) {
  const int tid  = threadIdx.x;
  const int wv   = tid >> 5;                    // wave in block
  const int lane = tid & 31;
  const int q0   = blockIdx.x * 64 + wv * 16;   // this wave's q tile
  const int bh   = blockIdx.y;
  const int b    = bh >> 2;   // / H_
  const int h    = bh & 3;
  const int half = lane >> 4;
  const int lrow = lane & 15;
  const int c0   = half ? 8 : 0;
  const float NEG_INF = -__builtin_inff();

  __shared__ __align__(16) __bf16 Ps[4][16][32];  // per-wave slice

  // resident Q fragments: 8 x (16x32 bf16) covering DK=256 (pre-scaled)
  ABFrag qf[8];
  const __bf16* qrow = QP + (size_t)(b * L_ + q0 + lrow) * D_ + h * DK_;
#pragma unroll
  for (int dt = 0; dt < 8; ++dt) {
    qf[dt].h[0] = *(const v8bf*)(qrow + dt * 32 + c0);
    qf[dt].h[1] = *(const v8bf*)(qrow + dt * 32 + c0 + 16);
  }

  v8f oacc[16];
#pragma unroll
  for (int nt = 0; nt < 16; ++nt) oacc[nt] = vzero8();
  float mrow[8], lsum[8];
#pragma unroll
  for (int r = 0; r < 8; ++r) { mrow[r] = NEG_INF; lsum[r] = 0.f; }

  const unsigned char* mbase_p = mask + (size_t)b * L_ * L_;

  for (int j0 = 0; j0 < L_; j0 += 32) {
    // ---- S = Q * K^T : two 16x16 tiles, 8 d-steps each ----
    v8f s[2];
    s[0] = vzero8(); s[1] = vzero8();
#pragma unroll
    for (int jt = 0; jt < 2; ++jt) {
      const __bf16* krow =
          KP + (size_t)(b * L_ + j0 + jt * 16 + lrow) * D_ + h * DK_;
#pragma unroll
      for (int dt = 0; dt < 8; ++dt) {
        ABFrag kf;
        kf.h[0] = *(const v8bf*)(krow + dt * 32 + c0);
        kf.h[1] = *(const v8bf*)(krow + dt * 32 + c0 + 16);
        s[jt] = wmma_bf16(qf[dt].v, kf.v, s[jt]);
      }
    }
    // ---- mask (scores already scaled via Q) ----
    float sv[2][8];
#pragma unroll
    for (int jt = 0; jt < 2; ++jt) {
      const int j = j0 + jt * 16 + lrow;
#pragma unroll
      for (int r = 0; r < 8; ++r) {
        const int mq = q0 + half * 8 + r;
        float val = s[jt][r];
        if (mbase_p[(size_t)mq * L_ + j]) val = NEG_INF;
        sv[jt][r] = val;
      }
    }
    // ---- online softmax (C-layout row: vgpr r, 16 lanes of one half) ----
#pragma unroll
    for (int r = 0; r < 8; ++r) {
      float mx = fmaxf(sv[0][r], sv[1][r]);
#pragma unroll
      for (int off = 8; off >= 1; off >>= 1)   // stays inside 16-lane half
        mx = fmaxf(mx, __shfl_xor(mx, off, 32));
      const float mnew  = fmaxf(mrow[r], mx);
      const float alpha = (mrow[r] == NEG_INF) ? 0.f : __expf(mrow[r] - mnew);
      float ps = 0.f;
#pragma unroll
      for (int jt = 0; jt < 2; ++jt) {
        float p = (sv[jt][r] == NEG_INF) ? 0.f : __expf(sv[jt][r] - mnew);
        sv[jt][r] = p;
        ps += p;
      }
#pragma unroll
      for (int off = 8; off >= 1; off >>= 1)
        ps += __shfl_xor(ps, off, 32);
      lsum[r] = lsum[r] * alpha + ps;
      mrow[r] = mnew;
#pragma unroll
      for (int nt = 0; nt < 16; ++nt) oacc[nt][r] *= alpha;
    }
    // ---- P (C layout) -> private LDS slice -> A layout ----
    // Same-wave ds_store -> ds_load is ordered by hardware (DS in-order/wave).
#pragma unroll
    for (int jt = 0; jt < 2; ++jt)
#pragma unroll
      for (int r = 0; r < 8; ++r)
        Ps[wv][half * 8 + r][jt * 16 + lrow] = f2bf(sv[jt][r]);
    ABFrag pf;
    pf.h[0] = *(const v8bf*)&Ps[wv][lrow][c0];
    pf.h[1] = *(const v8bf*)&Ps[wv][lrow][c0 + 16];
    // ---- O += P * V : 16 N-tiles of 16 over DK ----
#pragma unroll
    for (int nt = 0; nt < 16; ++nt) {
      const __bf16* vcol =
          VPT + ((size_t)(b * D_ + h * DK_ + nt * 16 + lrow)) * L_ + j0;
      ABFrag vf;
      vf.h[0] = *(const v8bf*)(vcol + c0);
      vf.h[1] = *(const v8bf*)(vcol + c0 + 16);
      oacc[nt] = wmma_bf16(pf.v, vf.v, oacc[nt]);
    }
  }

  // ---- normalize by l (0 for fully-masked rows) and store bf16 ----
#pragma unroll
  for (int r = 0; r < 8; ++r) {
    const float rl = (lsum[r] > 0.f) ? (1.0f / lsum[r]) : 0.f;
    const int mq = q0 + half * 8 + r;
    __bf16* orow = O + (size_t)(b * L_ + mq) * D_ + h * DK_;
#pragma unroll
    for (int nt = 0; nt < 16; ++nt)
      orow[nt * 16 + lrow] = f2bf(oacc[nt][r] * rl);
  }
}

// ---------------------------------------------------------------------------
// Kernel 3a: X = O @ Wfc^T + bfc + residual(q)   (f32 out to workspace)
// ---------------------------------------------------------------------------
__global__ __launch_bounds__(256) void fc_residual_kernel(
    const __bf16* __restrict__ Obf, const float* __restrict__ Wfc,
    const float* __restrict__ bfc, const float* __restrict__ qres,
    float* __restrict__ X) {
  const int m0   = blockIdx.y * 128;
  const int n0   = blockIdx.x * 64;
  const int tid  = threadIdx.x;
  const int wave = tid >> 5;
  const int lane = tid & 31;
  const int half = lane >> 4;
  const int lrow = lane & 15;
  const int wm   = wave & 3;
  const int wn   = wave >> 2;
  const int c0   = half ? 8 : 0;

  __shared__ __align__(16) __bf16 As[128][64];
  __shared__ __align__(16) __bf16 Bs[64][64];

  v8f acc[2][2];
#pragma unroll
  for (int mr = 0; mr < 2; ++mr)
#pragma unroll
    for (int nr = 0; nr < 2; ++nr) acc[mr][nr] = vzero8();

  for (int k0 = 0; k0 < D_; k0 += 64) {
    // A tile already bf16: 4 x v8bf copies per thread (8192 elems/tile)
#pragma unroll
    for (int i = 0; i < 4; ++i) {
      const int e = (tid + i * 256) * 8;
      const int r = e >> 6, c = e & 63;
      *(v8bf*)&As[r][c] =
          *(const v8bf*)&Obf[(size_t)(m0 + r) * D_ + k0 + c];
    }
    // W tile: vectorized f32 load + native bf16 pack
#pragma unroll
    for (int i = 0; i < 2; ++i) {
      const int e = (tid + i * 256) * 8;
      const int r = e >> 6, c = e & 63;
      const v4f* src = (const v4f*)&Wfc[(size_t)(n0 + r) * D_ + k0 + c];
      *(v8bf*)&Bs[r][c] = pack8(src[0], src[1]);
    }
    if (k0 + 64 < D_)
      __builtin_prefetch(&Wfc[(size_t)(n0 + (tid >> 2)) * D_ + k0 + 64], 0, 1);
    __syncthreads();

#pragma unroll
    for (int ks = 0; ks < 2; ++ks) {
      const int kc = ks * 32 + c0;
      ABFrag af[2], bf_[2];
#pragma unroll
      for (int mr = 0; mr < 2; ++mr) {
        const int ar = wm * 32 + mr * 16 + lrow;
        af[mr].h[0] = *(const v8bf*)&As[ar][kc];
        af[mr].h[1] = *(const v8bf*)&As[ar][kc + 16];
      }
#pragma unroll
      for (int nr = 0; nr < 2; ++nr) {
        const int br = wn * 32 + nr * 16 + lrow;
        bf_[nr].h[0] = *(const v8bf*)&Bs[br][kc];
        bf_[nr].h[1] = *(const v8bf*)&Bs[br][kc + 16];
      }
#pragma unroll
      for (int mr = 0; mr < 2; ++mr)
#pragma unroll
        for (int nr = 0; nr < 2; ++nr)
          acc[mr][nr] = wmma_bf16(af[mr].v, bf_[nr].v, acc[mr][nr]);
    }
    __syncthreads();
  }

#pragma unroll
  for (int nr = 0; nr < 2; ++nr) {
    const int n = n0 + wn * 32 + nr * 16 + lrow;
    const float bb = bfc[n];
#pragma unroll
    for (int mr = 0; mr < 2; ++mr) {
      const int mbase = m0 + wm * 32 + mr * 16 + half * 8;
#pragma unroll
      for (int r = 0; r < 8; ++r) {
        const int m = mbase + r;
        X[(size_t)m * D_ + n] =
            acc[mr][nr][r] + bb + qres[(size_t)m * D_ + n];
      }
    }
  }
}

// ---------------------------------------------------------------------------
// Kernel 3b: LayerNorm over D=1024 per row. 256 threads/row, b128 I/O.
// ---------------------------------------------------------------------------
__global__ __launch_bounds__(256) void layernorm_kernel(
    const float* __restrict__ X, const float* __restrict__ gamma,
    const float* __restrict__ beta, float* __restrict__ out) {
  const int row = blockIdx.x;
  const int tid = threadIdx.x;
  const float* x = X + (size_t)row * D_;

  const v4f xv = *(const v4f*)&x[tid * 4];
  float s = 0.f, sq = 0.f;
#pragma unroll
  for (int i = 0; i < 4; ++i) { s += xv[i]; sq += xv[i] * xv[i]; }
#pragma unroll
  for (int off = 16; off >= 1; off >>= 1) {
    s  += __shfl_xor(s, off, 32);
    sq += __shfl_xor(sq, off, 32);
  }
  __shared__ float red_s[8], red_q[8];
  const int wave = tid >> 5, lane = tid & 31;
  if (lane == 0) { red_s[wave] = s; red_q[wave] = sq; }
  __syncthreads();
  if (wave == 0) {
    float a  = (lane < 8) ? red_s[lane] : 0.f;
    float b2 = (lane < 8) ? red_q[lane] : 0.f;
#pragma unroll
    for (int off = 4; off >= 1; off >>= 1) {
      a  += __shfl_xor(a, off, 32);
      b2 += __shfl_xor(b2, off, 32);
    }
    if (lane == 0) { red_s[0] = a; red_q[0] = b2; }
  }
  __syncthreads();
  const float mean = red_s[0] * (1.0f / D_);
  const float var  = red_q[0] * (1.0f / D_) - mean * mean;
  const float rstd = rsqrtf(var + EPS_);

  const v4f g = *(const v4f*)&gamma[tid * 4];
  const v4f be = *(const v4f*)&beta[tid * 4];
  v4f o;
#pragma unroll
  for (int i = 0; i < 4; ++i)
    o[i] = (xv[i] - mean) * rstd * g[i] + be[i];
  *(v4f*)&out[(size_t)row * D_ + tid * 4] = o;
}

// ---------------------------------------------------------------------------
// Launcher
// ---------------------------------------------------------------------------
extern "C" void kernel_launch(void* const* d_in, const int* in_sizes, int n_in,
                              void* d_out, int out_size, void* d_ws,
                              size_t ws_size, hipStream_t stream) {
  const float* q    = (const float*)d_in[0];
  const float* k    = (const float*)d_in[1];
  const float* v    = (const float*)d_in[2];
  const unsigned char* mask = (const unsigned char*)d_in[3];  // bool
  const float* Wq   = (const float*)d_in[4];
  const float* bq   = (const float*)d_in[5];
  const float* Wk   = (const float*)d_in[6];
  const float* bk   = (const float*)d_in[7];
  const float* Wv   = (const float*)d_in[8];
  const float* bv   = (const float*)d_in[9];
  const float* Wfc  = (const float*)d_in[10];
  const float* bfc  = (const float*)d_in[11];
  const float* gamma = (const float*)d_in[12];
  const float* beta  = (const float*)d_in[13];

  char* ws = (char*)d_ws;
  const size_t SZ = (size_t)M_ * D_;  // elements per activation tensor
  __bf16* QP  = (__bf16*)(ws);                 // [M, D] bf16 (pre-scaled)
  __bf16* KP  = (__bf16*)(ws + SZ * 2);        // [M, D] bf16
  __bf16* VPT = (__bf16*)(ws + SZ * 4);        // [B, D, L] bf16 (transposed)
  __bf16* Obf = (__bf16*)(ws + SZ * 6);        // [M, D] bf16
  float*  X   = (float*)(ws);                  // [M, D] f32 (reuses QP+KP)

  qkv_proj_kernel<<<dim3(D_ / 64, M_ / 128, 3), 256, 0, stream>>>(
      q, k, v, Wq, bq, Wk, bk, Wv, bv, QP, KP, VPT);
  flash_attn_kernel<<<dim3(L_ / 64, B_ * H_), 128, 0, stream>>>(
      QP, KP, VPT, mask, Obf);
  fc_residual_kernel<<<dim3(D_ / 64, M_ / 128), 256, 0, stream>>>(
      Obf, Wfc, bfc, q, X);
  layernorm_kernel<<<dim3(M_), 256, 0, stream>>>(X, gamma, beta,
                                                 (float*)d_out);
}
